// CustomSpatialTemporalTransformer_76759655514990
// MI455X (gfx1250) — compile-verified, matrix operand order checked by reading, not run on hardware
//
#include <hip/hip_runtime.h>

// ---------------------------------------------------------------------------
// CustomSpatialTemporalTransformer forward for MI455X (gfx1250, wave32, WMMA)
// All GEMMs (patch-embed, QKV, Q*K^T, P*V, Wo, FF1, FF2, inverse-proj) run on
// v_wmma_f32_16x16x32_f16. GEMM uses a 32x32 per-wave register tile (2x2
// fragments) so each A/B fragment feeds two WMMAs (2 b128 loads per WMMA).
// Attention is flash-style (online softmax), mask evaluated analytically.
// ---------------------------------------------------------------------------

#define NTOK  2304   // S*P tokens
#define PNUM  576    // patches per frame
#define EMB   512    // d_model
#define FFD   2048   // feedforward
#define PDIM  2048   // patch dim
#define NLAYER 6
#define NHEAD  8     // dh = 64

typedef __attribute__((ext_vector_type(16))) _Float16 v16h;
typedef __attribute__((ext_vector_type(8)))  _Float16 v8h;
typedef __attribute__((ext_vector_type(8)))  float    v8f;

union frag16 { v16h v; v8h p[2]; };

__device__ __forceinline__ int iabs_(int x) { return x < 0 ? -x : x; }

// --------------------------- weight transpose ------------------------------
// in: f32 [K][Ncol] row-major  ->  out: f16 [Ncol][K]  (B^T for WMMA B loads)
__global__ void wT_kernel(const float* __restrict__ in, _Float16* __restrict__ out,
                          int K, int Ncol) {
  long idx = (long)blockIdx.x * blockDim.x + threadIdx.x;
  if (idx >= (long)K * Ncol) return;
  int k = (int)(idx / Ncol), n = (int)(idx % Ncol);
  out[(long)n * K + k] = (_Float16)in[idx];
}

// --------------------------- patch gather ----------------------------------
// 'b s (pc nc) (nh ph) (nw pw) -> (s nc nh nw) (ph pw pc)' as f16 [NTOK][PDIM]
__global__ void patch_gather(const float* __restrict__ in, _Float16* __restrict__ out) {
  long idx = (long)blockIdx.x * blockDim.x + threadIdx.x;
  if (idx >= (long)NTOK * PDIM) return;
  int t = (int)(idx / PDIM), f = (int)(idx % PDIM);
  int s = t / PNUM, p = t % PNUM;
  int nc = p / 144, nh = (p / 12) % 12, nw = p % 12;
  int pc = f & 31, pw = (f >> 5) & 7, ph = f >> 8;
  int c = pc * 4 + nc, hh = nh * 8 + ph, ww = nw * 8 + pw;
  out[idx] = (_Float16)in[(((long)s * 128 + c) * 96 + hh) * 96 + ww];
}

// --------------------------- generic WMMA GEMM -----------------------------
// C[M][Nc] = A[M][K] (f16, lda==K) * BT[Nc][K] (f16) + bias; optional relu;
// writes f32 and/or f16. One wave computes a 32x32 tile (2x2 fragments);
// 8 waves per block cover 32 rows x 256 cols. M%32==0, Nc%256==0, K%32==0.
__global__ void __launch_bounds__(256) wmma_gemm(
    const _Float16* __restrict__ A, const _Float16* __restrict__ BT,
    const float* __restrict__ bias, float* __restrict__ Cf,
    _Float16* __restrict__ Ch, int M, int Nc, int K, int relu) {
  const int wave = threadIdx.x >> 5;
  const int lane = threadIdx.x & 31;
  const int half = lane >> 4;
  const int ln16 = lane & 15;
  const int tileM = blockIdx.y * 32;
  const int tileN = (blockIdx.x * 8 + wave) * 32;
  if (tileN >= Nc) return;  // wave-uniform

  const _Float16* arow0 = A + (long)(tileM + ln16) * K;
  const _Float16* arow1 = A + (long)(tileM + 16 + ln16) * K;
  const _Float16* brow0 = BT + (long)(tileN + ln16) * K;
  const _Float16* brow1 = BT + (long)(tileN + 16 + ln16) * K;
  v8f acc00 = {}, acc01 = {}, acc10 = {}, acc11 = {};
  for (int k0 = 0; k0 < K; k0 += 32) {
    frag16 a0, a1, b0, b1;
    a0.p[0] = *(const v8h*)(arow0 + k0 + 8 * half);
    a0.p[1] = *(const v8h*)(arow0 + k0 + 16 + 8 * half);
    a1.p[0] = *(const v8h*)(arow1 + k0 + 8 * half);
    a1.p[1] = *(const v8h*)(arow1 + k0 + 16 + 8 * half);
    b0.p[0] = *(const v8h*)(brow0 + k0 + 8 * half);
    b0.p[1] = *(const v8h*)(brow0 + k0 + 16 + 8 * half);
    b1.p[0] = *(const v8h*)(brow1 + k0 + 8 * half);
    b1.p[1] = *(const v8h*)(brow1 + k0 + 16 + 8 * half);
    acc00 = __builtin_amdgcn_wmma_f32_16x16x32_f16(false, a0.v, false, b0.v, (short)0, acc00, false, false);
    acc01 = __builtin_amdgcn_wmma_f32_16x16x32_f16(false, a0.v, false, b1.v, (short)0, acc01, false, false);
    acc10 = __builtin_amdgcn_wmma_f32_16x16x32_f16(false, a1.v, false, b0.v, (short)0, acc10, false, false);
    acc11 = __builtin_amdgcn_wmma_f32_16x16x32_f16(false, a1.v, false, b1.v, (short)0, acc11, false, false);
  }
  float bb0 = bias ? bias[tileN + ln16] : 0.0f;
  float bb1 = bias ? bias[tileN + 16 + ln16] : 0.0f;
#pragma unroll
  for (int r = 0; r < 8; ++r) {
    int row0 = tileM + r + 8 * half;
    int row1 = tileM + 16 + r + 8 * half;
    float v00 = acc00[r] + bb0, v01 = acc01[r] + bb1;
    float v10 = acc10[r] + bb0, v11 = acc11[r] + bb1;
    if (relu) {
      v00 = fmaxf(v00, 0.0f); v01 = fmaxf(v01, 0.0f);
      v10 = fmaxf(v10, 0.0f); v11 = fmaxf(v11, 0.0f);
    }
    long i00 = (long)row0 * Nc + tileN + ln16;
    long i01 = (long)row0 * Nc + tileN + 16 + ln16;
    long i10 = (long)row1 * Nc + tileN + ln16;
    long i11 = (long)row1 * Nc + tileN + 16 + ln16;
    if (Cf) { Cf[i00] = v00; Cf[i01] = v01; Cf[i10] = v10; Cf[i11] = v11; }
    if (Ch) {
      Ch[i00] = (_Float16)v00; Ch[i01] = (_Float16)v01;
      Ch[i10] = (_Float16)v10; Ch[i11] = (_Float16)v11;
    }
  }
}

// --------------------------- embed add -------------------------------------
__global__ void embed_add(const float* __restrict__ xpe, const float* __restrict__ pos,
                          const float* __restrict__ temp, float* __restrict__ xf,
                          _Float16* __restrict__ xh) {
  long idx = (long)blockIdx.x * blockDim.x + threadIdx.x;
  if (idx >= (long)NTOK * EMB) return;
  int t = (int)(idx / EMB), e = (int)(idx % EMB);
  int s = t / PNUM, p = t % PNUM;
  float v = xpe[idx] + pos[(long)p * EMB + e] + temp[(long)s * EMB + e];
  xf[idx] = v;
  xh[idx] = (_Float16)v;
}

// --------------------------- V transpose per head --------------------------
// qkv f16 [NTOK][1536] -> vT f16 [EMB][NTOK] (row = h*64+d)
__global__ void v_transpose(const _Float16* __restrict__ qkv, _Float16* __restrict__ vT) {
  long idx = (long)blockIdx.x * blockDim.x + threadIdx.x;
  if (idx >= (long)NTOK * EMB) return;
  int t = (int)(idx / EMB), e = (int)(idx % EMB);
  vT[(long)e * NTOK + t] = qkv[(long)t * 1536 + 1024 + e];
}

// --------------------------- flash attention -------------------------------
// grid (NHEAD, NTOK/16/8), block 256. One wave: one head x one 16-query tile.
__global__ void __launch_bounds__(256) attn_kernel(
    const _Float16* __restrict__ qkv, const _Float16* __restrict__ vT,
    _Float16* __restrict__ obuf) {
  __shared__ _Float16 stage[8][512];  // per-wave 16x32 f16 P-staging
  const int wave = threadIdx.x >> 5;
  const int lane = threadIdx.x & 31;
  const int half = lane >> 4;
  const int ln16 = lane & 15;
  const int hd = blockIdx.x;
  const int qtile = blockIdx.y * 8 + wave;
  _Float16* st = stage[wave];

  // Q fragments: rows qtile*16+m, K-dim = dh (two 32-chunks)
  const _Float16* qbase = qkv + (long)(qtile * 16 + ln16) * 1536 + hd * 64;
  frag16 qa0, qa1;
  qa0.p[0] = *(const v8h*)(qbase + 8 * half);
  qa0.p[1] = *(const v8h*)(qbase + 16 + 8 * half);
  qa1.p[0] = *(const v8h*)(qbase + 32 + 8 * half);
  qa1.p[1] = *(const v8h*)(qbase + 48 + 8 * half);

  int qc[8], qh[8], qw[8];
#pragma unroll
  for (int r = 0; r < 8; ++r) {
    int q = qtile * 16 + r + 8 * half;
    int p = q % PNUM;
    qc[r] = p / 144; qh[r] = (p / 12) % 12; qw[r] = p % 12;
  }

  float mrun[8], lrun[8];
  v8f o0 = {}, o1 = {}, o2 = {}, o3 = {};
#pragma unroll
  for (int r = 0; r < 8; ++r) { mrun[r] = -1e30f; lrun[r] = 0.0f; }
  const float scale = 0.125f;  // 1/sqrt(64)

  for (int kt2 = 0; kt2 < NTOK / 32; ++kt2) {
    const int kb2 = kt2 * 32;
    v8f s0 = {}, s1 = {};
    {
      const _Float16* kb = qkv + (long)(kb2 + ln16) * 1536 + EMB + hd * 64;
      frag16 k0, k1;
      k0.p[0] = *(const v8h*)(kb + 8 * half);      k0.p[1] = *(const v8h*)(kb + 16 + 8 * half);
      k1.p[0] = *(const v8h*)(kb + 32 + 8 * half); k1.p[1] = *(const v8h*)(kb + 48 + 8 * half);
      s0 = __builtin_amdgcn_wmma_f32_16x16x32_f16(false, qa0.v, false, k0.v, (short)0, s0, false, false);
      s0 = __builtin_amdgcn_wmma_f32_16x16x32_f16(false, qa1.v, false, k1.v, (short)0, s0, false, false);
    }
    {
      const _Float16* kb = qkv + (long)(kb2 + 16 + ln16) * 1536 + EMB + hd * 64;
      frag16 k0, k1;
      k0.p[0] = *(const v8h*)(kb + 8 * half);      k0.p[1] = *(const v8h*)(kb + 16 + 8 * half);
      k1.p[0] = *(const v8h*)(kb + 32 + 8 * half); k1.p[1] = *(const v8h*)(kb + 48 + 8 * half);
      s1 = __builtin_amdgcn_wmma_f32_16x16x32_f16(false, qa0.v, false, k0.v, (short)0, s1, false, false);
      s1 = __builtin_amdgcn_wmma_f32_16x16x32_f16(false, qa1.v, false, k1.v, (short)0, s1, false, false);
    }
    // local-neighborhood mask from patch-grid coords (tiled over time)
    int kp0 = (kb2 + ln16) % PNUM;
    int kc0 = kp0 / 144, kh0 = (kp0 / 12) % 12, kw0 = kp0 % 12;
    int kp1 = (kb2 + 16 + ln16) % PNUM;
    int kc1 = kp1 / 144, kh1 = (kp1 / 12) % 12, kw1 = kp1 % 12;
#pragma unroll
    for (int r = 0; r < 8; ++r) {
      float a0 = s0[r] * scale;
      float a1 = s1[r] * scale;
      bool ok0 = iabs_(qc[r] - kc0) <= 1 && iabs_(qh[r] - kh0) <= 2 && iabs_(qw[r] - kw0) <= 2;
      bool ok1 = iabs_(qc[r] - kc1) <= 1 && iabs_(qh[r] - kh1) <= 2 && iabs_(qw[r] - kw1) <= 2;
      a0 = ok0 ? a0 : -1e30f;
      a1 = ok1 ? a1 : -1e30f;
      float rmax = fmaxf(a0, a1);
#pragma unroll
      for (int off = 1; off < 16; off <<= 1) rmax = fmaxf(rmax, __shfl_xor(rmax, off));
      float nm = fmaxf(mrun[r], rmax);
      float p0 = __expf(a0 - nm);
      float p1 = __expf(a1 - nm);
      float rs = p0 + p1;
#pragma unroll
      for (int off = 1; off < 16; off <<= 1) rs += __shfl_xor(rs, off);
      float al = __expf(mrun[r] - nm);
      lrun[r] = lrun[r] * al + rs;
      mrun[r] = nm;
      o0[r] *= al; o1[r] *= al; o2[r] *= al; o3[r] *= al;
      st[(r + 8 * half) * 32 + ln16] = (_Float16)p0;
      st[(r + 8 * half) * 32 + 16 + ln16] = (_Float16)p1;
    }
    // P: D-layout -> A-fragment via LDS (same-wave LDS is in-order)
    frag16 pa;
    pa.p[0] = *(const v8h*)(st + ln16 * 32 + 8 * half);
    pa.p[1] = *(const v8h*)(st + ln16 * 32 + 16 + 8 * half);
    {
      frag16 vb;
      const _Float16* v0 = vT + (long)(hd * 64 + 0 + ln16) * NTOK + kb2;
      vb.p[0] = *(const v8h*)(v0 + 8 * half); vb.p[1] = *(const v8h*)(v0 + 16 + 8 * half);
      o0 = __builtin_amdgcn_wmma_f32_16x16x32_f16(false, pa.v, false, vb.v, (short)0, o0, false, false);
      const _Float16* v1 = vT + (long)(hd * 64 + 16 + ln16) * NTOK + kb2;
      vb.p[0] = *(const v8h*)(v1 + 8 * half); vb.p[1] = *(const v8h*)(v1 + 16 + 8 * half);
      o1 = __builtin_amdgcn_wmma_f32_16x16x32_f16(false, pa.v, false, vb.v, (short)0, o1, false, false);
      const _Float16* v2 = vT + (long)(hd * 64 + 32 + ln16) * NTOK + kb2;
      vb.p[0] = *(const v8h*)(v2 + 8 * half); vb.p[1] = *(const v8h*)(v2 + 16 + 8 * half);
      o2 = __builtin_amdgcn_wmma_f32_16x16x32_f16(false, pa.v, false, vb.v, (short)0, o2, false, false);
      const _Float16* v3 = vT + (long)(hd * 64 + 48 + ln16) * NTOK + kb2;
      vb.p[0] = *(const v8h*)(v3 + 8 * half); vb.p[1] = *(const v8h*)(v3 + 16 + 8 * half);
      o3 = __builtin_amdgcn_wmma_f32_16x16x32_f16(false, pa.v, false, vb.v, (short)0, o3, false, false);
    }
  }
#pragma unroll
  for (int r = 0; r < 8; ++r) {
    float inv = 1.0f / lrun[r];
    long row = qtile * 16 + r + 8 * half;
    _Float16* ob = obuf + row * EMB + hd * 64 + ln16;
    ob[0]  = (_Float16)(o0[r] * inv);
    ob[16] = (_Float16)(o1[r] * inv);
    ob[32] = (_Float16)(o2[r] * inv);
    ob[48] = (_Float16)(o3[r] * inv);
  }
}

// --------------------------- residual + LayerNorm --------------------------
// x = LN(xres + delta); writes f32 (residual stream) and f16 (next GEMM A).
__global__ void __launch_bounds__(256) resid_ln(
    const float* __restrict__ xres, const float* __restrict__ delta,
    const float* __restrict__ w, const float* __restrict__ b,
    float* __restrict__ xout, _Float16* __restrict__ hout) {
  __shared__ float red[256];
  int t = blockIdx.x, tid = threadIdx.x;
  long base = (long)t * EMB;
  float v0 = xres[base + tid] + delta[base + tid];
  float v1 = xres[base + 256 + tid] + delta[base + 256 + tid];
  red[tid] = v0 + v1;
  __syncthreads();
  for (int s = 128; s > 0; s >>= 1) { if (tid < s) red[tid] += red[tid + s]; __syncthreads(); }
  float mean = red[0] * (1.0f / EMB);
  __syncthreads();
  float d0 = v0 - mean, d1 = v1 - mean;
  red[tid] = d0 * d0 + d1 * d1;
  __syncthreads();
  for (int s = 128; s > 0; s >>= 1) { if (tid < s) red[tid] += red[tid + s]; __syncthreads(); }
  float inv = rsqrtf(red[0] * (1.0f / EMB) + 1e-5f);
  float y0 = d0 * inv * w[tid] + b[tid];
  float y1 = d1 * inv * w[256 + tid] + b[256 + tid];
  xout[base + tid] = y0;        xout[base + 256 + tid] = y1;
  hout[base + tid] = (_Float16)y0; hout[base + 256 + tid] = (_Float16)y1;
}

// --------------------------- un-patch --------------------------------------
__global__ void unpatch(const float* __restrict__ y, float* __restrict__ out) {
  long idx = (long)blockIdx.x * blockDim.x + threadIdx.x;
  if (idx >= (long)PNUM * PDIM) return;
  int p = (int)(idx / PDIM), f = (int)(idx % PDIM);
  int nc = p / 144, nh = (p / 12) % 12, nw = p % 12;
  int pc = f & 31, pw = (f >> 5) & 7, ph = f >> 8;
  int c = pc * 4 + nc, hh = nh * 8 + ph, ww = nw * 8 + pw;
  out[((long)c * 96 + hh) * 96 + ww] = y[idx];
}

// ---------------------------------------------------------------------------
extern "C" void kernel_launch(void* const* d_in, const int* in_sizes, int n_in,
                              void* d_out, int out_size, void* d_ws, size_t ws_size,
                              hipStream_t stream) {
  (void)in_sizes; (void)n_in; (void)out_size; (void)ws_size;
  const float* emb  = (const float*)d_in[0];
  const float* Wp   = (const float*)d_in[1];
  const float* bp   = (const float*)d_in[2];
  const float* pos  = (const float*)d_in[3];
  const float* temp = (const float*)d_in[4];
  const float* Wqkv = (const float*)d_in[5];
  const float* bqkv = (const float*)d_in[6];
  const float* Wo   = (const float*)d_in[7];
  const float* bo   = (const float*)d_in[8];
  const float* W1   = (const float*)d_in[9];
  const float* b1   = (const float*)d_in[10];
  const float* W2   = (const float*)d_in[11];
  const float* b2   = (const float*)d_in[12];
  const float* ln1w = (const float*)d_in[13];
  const float* ln1b = (const float*)d_in[14];
  const float* ln2w = (const float*)d_in[15];
  const float* ln2b = (const float*)d_in[16];
  const float* Winv = (const float*)d_in[17];
  const float* binv = (const float*)d_in[18];

  char* ws = (char*)d_ws;
  size_t off = 0;
  auto alloc = [&](size_t bytes) -> void* {
    void* p = ws + off;
    off += (bytes + 255) & ~(size_t)255;
    return p;
  };
  _Float16* wpT   = (_Float16*)alloc((size_t)PDIM * EMB * 2);
  _Float16* wqT   = (_Float16*)alloc((size_t)NLAYER * 3 * EMB * EMB * 2);
  _Float16* woT   = (_Float16*)alloc((size_t)NLAYER * EMB * EMB * 2);
  _Float16* w1T   = (_Float16*)alloc((size_t)NLAYER * FFD * EMB * 2);
  _Float16* w2T   = (_Float16*)alloc((size_t)NLAYER * EMB * FFD * 2);
  _Float16* wiT   = (_Float16*)alloc((size_t)EMB * PDIM * 2);
  _Float16* xp16  = (_Float16*)alloc((size_t)NTOK * PDIM * 2);
  float*    xpe   = (float*)   alloc((size_t)NTOK * EMB * 4);
  float*    xf    = (float*)   alloc((size_t)NTOK * EMB * 4);
  _Float16* x16   = (_Float16*)alloc((size_t)NTOK * EMB * 2);
  _Float16* qkv16 = (_Float16*)alloc((size_t)NTOK * 3 * EMB * 2);
  _Float16* vT    = (_Float16*)alloc((size_t)EMB * NTOK * 2);
  _Float16* o16   = (_Float16*)alloc((size_t)NTOK * EMB * 2);
  float*    dly   = (float*)   alloc((size_t)NTOK * EMB * 4);
  _Float16* h16   = (_Float16*)alloc((size_t)NTOK * FFD * 2);
  float*    yf    = (float*)   alloc((size_t)PNUM * PDIM * 4);

  auto gb = [](long n) { return dim3((unsigned)((n + 255) / 256)); };

  // Weight transposes (f32 -> f16 B^T)
  wT_kernel<<<gb((long)PDIM * EMB), 256, 0, stream>>>(Wp, wpT, PDIM, EMB);
  for (int l = 0; l < NLAYER; ++l) {
    wT_kernel<<<gb((long)EMB * 3 * EMB), 256, 0, stream>>>(
        Wqkv + (long)l * EMB * 3 * EMB, wqT + (long)l * 3 * EMB * EMB, EMB, 3 * EMB);
    wT_kernel<<<gb((long)EMB * EMB), 256, 0, stream>>>(
        Wo + (long)l * EMB * EMB, woT + (long)l * EMB * EMB, EMB, EMB);
    wT_kernel<<<gb((long)EMB * FFD), 256, 0, stream>>>(
        W1 + (long)l * EMB * FFD, w1T + (long)l * FFD * EMB, EMB, FFD);
    wT_kernel<<<gb((long)FFD * EMB), 256, 0, stream>>>(
        W2 + (long)l * FFD * EMB, w2T + (long)l * EMB * FFD, FFD, EMB);
  }
  wT_kernel<<<gb((long)EMB * PDIM), 256, 0, stream>>>(Winv, wiT, EMB, PDIM);

  // Patch gather + embed
  patch_gather<<<gb((long)NTOK * PDIM), 256, 0, stream>>>(emb, xp16);
  wmma_gemm<<<dim3(EMB / 256, NTOK / 32), 256, 0, stream>>>(
      xp16, wpT, bp, xpe, (_Float16*)nullptr, NTOK, EMB, PDIM, 0);
  embed_add<<<gb((long)NTOK * EMB), 256, 0, stream>>>(xpe, pos, temp, xf, x16);

  for (int l = 0; l < NLAYER; ++l) {
    wmma_gemm<<<dim3(3 * EMB / 256, NTOK / 32), 256, 0, stream>>>(
        x16, wqT + (long)l * 3 * EMB * EMB, bqkv + (long)l * 3 * EMB,
        (float*)nullptr, qkv16, NTOK, 3 * EMB, EMB, 0);
    v_transpose<<<gb((long)NTOK * EMB), 256, 0, stream>>>(qkv16, vT);
    attn_kernel<<<dim3(NHEAD, NTOK / 16 / 8), 256, 0, stream>>>(qkv16, vT, o16);
    wmma_gemm<<<dim3(EMB / 256, NTOK / 32), 256, 0, stream>>>(
        o16, woT + (long)l * EMB * EMB, bo + (long)l * EMB,
        dly, (_Float16*)nullptr, NTOK, EMB, EMB, 0);
    resid_ln<<<NTOK, 256, 0, stream>>>(xf, dly, ln1w + (long)l * EMB,
                                       ln1b + (long)l * EMB, xf, x16);
    wmma_gemm<<<dim3(FFD / 256, NTOK / 32), 256, 0, stream>>>(
        x16, w1T + (long)l * FFD * EMB, b1 + (long)l * FFD,
        (float*)nullptr, h16, NTOK, FFD, EMB, 1);
    wmma_gemm<<<dim3(EMB / 256, NTOK / 32), 256, 0, stream>>>(
        h16, w2T + (long)l * EMB * FFD, b2 + (long)l * EMB,
        dly, (_Float16*)nullptr, NTOK, EMB, FFD, 0);
    resid_ln<<<NTOK, 256, 0, stream>>>(xf, dly, ln2w + (long)l * EMB,
                                       ln2b + (long)l * EMB, xf, x16);
  }

  // Last frame -> inverse projection -> un-patch
  wmma_gemm<<<dim3(PDIM / 256, PNUM / 32), 256, 0, stream>>>(
      x16 + (long)(NTOK - PNUM) * EMB, wiT, binv, yf, (_Float16*)nullptr,
      PNUM, PDIM, EMB, 0);
  unpatch<<<gb((long)PNUM * PDIM), 256, 0, stream>>>(yf, (float*)d_out);
}
